// AcceleratedAttention_55860344652083
// MI455X (gfx1250) — compile-verified, matrix operand order checked by reading, not run on hardware
//
#include <hip/hip_runtime.h>
#include <stdint.h>

typedef __attribute__((ext_vector_type(16))) __bf16 v16bf;
typedef __attribute__((ext_vector_type(8)))  float  v8f;

#define DIM      2048
#define SEQ      2048
#define NHEADS   32
#define NKVHEADS 8
#define HD       64
#define BATCH    2
#define MTOT     (BATCH * SEQ)                      // 4096
#define NQKV     ((NHEADS + 2 * NKVHEADS) * HD)     // 3072
#define SCALE    0.125f                             // 1/sqrt(64)
#define NEG_INF  (-1000000000.0f)

// ---------------------------------------------------------------------------
// helpers
// ---------------------------------------------------------------------------

__device__ __forceinline__ void lds_fence() {
    // order same-wave LDS producer lanes vs consumer lanes (DScnt drain)
    asm volatile("s_wait_dscnt 0" ::: "memory");
}

// Load one 16-bit A/B WMMA fragment (16x32 row-major along K) per the CDNA5
// VGPR layout: lane<16 holds K {0..7,16..23}, lane>=16 holds K {8..15,24..31}.
// Two contiguous 16-byte loads per lane.
__device__ __forceinline__ v16bf load_frag(const __bf16* __restrict__ base,
                                           int row, int ldk, int k0, int lane) {
    const __bf16* p = base + (size_t)row * ldk + k0 + ((lane >> 4) << 3);
    union { v16bf v; uint4 q[2]; } u;
    u.q[0] = *(const uint4*)(p);
    u.q[1] = *(const uint4*)(p + 16);
    return u.v;
}

#define WMMA_BF16(A, B, C) \
    __builtin_amdgcn_wmma_f32_16x16x32_bf16(false, (A), false, (B), (short)0, (C), false, false)

// ---------------------------------------------------------------------------
// f32 -> bf16 conversion
// ---------------------------------------------------------------------------
__global__ void f32_to_bf16_kernel(const float* __restrict__ in,
                                   __bf16* __restrict__ out, int n) {
    int i = blockIdx.x * blockDim.x + threadIdx.x;
    if (i < n) out[i] = (__bf16)in[i];
}

// ---------------------------------------------------------------------------
// GEMM1: qkv = x @ W_qkv^T  (M=4096, N=3072, K=2048), scatter into Q/K/V^T
// One wave computes a 32x64 tile: 8 accumulators, 8 WMMAs per 32-wide k-step
// (12 16-byte loads per 8 WMMAs -> 1.5 loads/WMMA of L2 traffic).
// ---------------------------------------------------------------------------
__global__ __launch_bounds__(128) void gemm_qkv_kernel(
        const __bf16* __restrict__ X, const __bf16* __restrict__ W,
        __bf16* __restrict__ Q, __bf16* __restrict__ Kc, __bf16* __restrict__ VT) {
    const int lane = threadIdx.x & 31;
    const int wave = (blockIdx.x * blockDim.x + threadIdx.x) >> 5;
    const int tilesN = NQKV / 64;                  // 48
    const int m0 = (wave / tilesN) * 32;
    const int n0 = (wave % tilesN) * 64;

    v8f acc[2][4];
    for (int r = 0; r < 2; ++r)
        for (int c = 0; c < 4; ++c)
            for (int i = 0; i < 8; ++i) acc[r][c][i] = 0.f;

    for (int k0 = 0; k0 < DIM; k0 += 32) {
        v16bf a0 = load_frag(X, m0 +      (lane & 15), DIM, k0, lane);
        v16bf a1 = load_frag(X, m0 + 16 + (lane & 15), DIM, k0, lane);
        v16bf b0 = load_frag(W, n0 +      (lane & 15), DIM, k0, lane);
        v16bf b1 = load_frag(W, n0 + 16 + (lane & 15), DIM, k0, lane);
        v16bf b2 = load_frag(W, n0 + 32 + (lane & 15), DIM, k0, lane);
        v16bf b3 = load_frag(W, n0 + 48 + (lane & 15), DIM, k0, lane);
        acc[0][0] = WMMA_BF16(a0, b0, acc[0][0]);
        acc[1][0] = WMMA_BF16(a1, b0, acc[1][0]);
        acc[0][1] = WMMA_BF16(a0, b1, acc[0][1]);
        acc[1][1] = WMMA_BF16(a1, b1, acc[1][1]);
        acc[0][2] = WMMA_BF16(a0, b2, acc[0][2]);
        acc[1][2] = WMMA_BF16(a1, b2, acc[1][2]);
        acc[0][3] = WMMA_BF16(a0, b3, acc[0][3]);
        acc[1][3] = WMMA_BF16(a1, b3, acc[1][3]);
    }

    const int half = (lane >> 4) << 3;             // 0 or 8
    for (int r = 0; r < 2; ++r)
        for (int c = 0; c < 4; ++c)
            for (int vg = 0; vg < 8; ++vg) {
                int m = m0 + 16 * r + vg + half;
                int n = n0 + 16 * c + (lane & 15);
                int b = m >> 11, s = m & (SEQ - 1);
                __bf16 bv = (__bf16)acc[r][c][vg];
                if (n < NHEADS * HD) {
                    int h = n >> 6, d = n & 63;
                    Q[(((size_t)b * NHEADS + h) * SEQ + s) * HD + d] = bv;
                } else if (n < (NHEADS + NKVHEADS) * HD) {
                    int nk = n - NHEADS * HD;
                    int h = nk >> 6, d = nk & 63;
                    Kc[(((size_t)b * NKVHEADS + h) * SEQ + s) * HD + d] = bv;
                } else {
                    int nv = n - (NHEADS + NKVHEADS) * HD;
                    int h = nv >> 6, d = nv & 63;
                    VT[(((size_t)b * NKVHEADS + h) * HD + d) * SEQ + s] = bv;
                }
            }
}

// ---------------------------------------------------------------------------
// Flash attention: one wave per (batch, head, 16-query tile).
// Online softmax over 32-key blocks; causal mask inline (kv head = h mod 8,
// matching jnp.tile in the reference).
// ---------------------------------------------------------------------------
__global__ __launch_bounds__(256) void attn_kernel(
        const __bf16* __restrict__ Q, const __bf16* __restrict__ Kc,
        const __bf16* __restrict__ VT, __bf16* __restrict__ AO) {
    __shared__ float  s_s[8][16][32];
    __shared__ __bf16 s_p[8][16][32];
    __shared__ float  s_stat[8][16];

    const int lane = threadIdx.x & 31;
    const int wv   = threadIdx.x >> 5;
    const int gw   = blockIdx.x * 8 + wv;          // 8192 waves total
    const int qt   = gw & 127;
    const int h    = (gw >> 7) & 31;
    const int b    = gw >> 12;
    const int q0   = qt * 16;
    const int kvh  = h & (NKVHEADS - 1);
    const int half = (lane >> 4) << 3;

    const __bf16* qbase = Q  + ((size_t)b * NHEADS   + h)   * SEQ * HD;
    const __bf16* kbase = Kc + ((size_t)b * NKVHEADS + kvh) * SEQ * HD;
    const __bf16* vbase = VT + ((size_t)b * NKVHEADS + kvh) * HD  * SEQ;

    // Q tile fragments (16x64 = two k-steps), loaded once
    v16bf qa0 = load_frag(qbase, q0 + (lane & 15), HD, 0,  lane);
    v16bf qa1 = load_frag(qbase, q0 + (lane & 15), HD, 32, lane);

    v8f oacc[4];
    for (int c = 0; c < 4; ++c)
        for (int i = 0; i < 8; ++i) oacc[c][i] = 0.f;
    float m_run = -1e30f, l_run = 0.f;

    const int nkb = (q0 + 47) >> 5;                // key blocks covering [0, q0+15]
    for (int kb = 0; kb < nkb; ++kb) {
        const int kb0 = kb * 32;

        // ---- S = Q K^T for 16q x 32k (4 WMMAs)
        v16bf k00 = load_frag(kbase, kb0 +      (lane & 15), HD, 0,  lane);
        v16bf k01 = load_frag(kbase, kb0 +      (lane & 15), HD, 32, lane);
        v16bf k10 = load_frag(kbase, kb0 + 16 + (lane & 15), HD, 0,  lane);
        v16bf k11 = load_frag(kbase, kb0 + 16 + (lane & 15), HD, 32, lane);
        v8f s0, s1;
        for (int i = 0; i < 8; ++i) { s0[i] = 0.f; s1[i] = 0.f; }
        s0 = WMMA_BF16(qa0, k00, s0);
        s0 = WMMA_BF16(qa1, k01, s0);
        s1 = WMMA_BF16(qa0, k10, s1);
        s1 = WMMA_BF16(qa1, k11, s1);

        // ---- spill scores to per-wave LDS (D-layout -> row-major)
        for (int vg = 0; vg < 8; ++vg) {
            int row = vg + half;
            s_s[wv][row][lane & 15]        = s0[vg];
            s_s[wv][row][16 + (lane & 15)] = s1[vg];
        }
        lds_fence();

        // ---- online softmax: lane r owns query row r
        if (lane < 16) {
            const int row = lane, qg = q0 + row;
            float mx = m_run;
            for (int j = 0; j < 32; ++j) {
                float s = s_s[wv][row][j] * SCALE;
                if (kb0 + j > qg) s = NEG_INF;
                s_s[wv][row][j] = s;
                mx = fmaxf(mx, s);
            }
            float alpha = __expf(m_run - mx);
            float lsum = 0.f;
            for (int j = 0; j < 32; ++j) {
                float e = __expf(s_s[wv][row][j] - mx);
                lsum += e;
                s_p[wv][row][j] = (__bf16)e;
            }
            l_run = l_run * alpha + lsum;
            m_run = mx;
            s_stat[wv][row] = alpha;
        }
        lds_fence();

        // ---- rescale O accumulators by per-row alpha
        for (int vg = 0; vg < 8; ++vg) {
            float a = s_stat[wv][vg + half];
            for (int c = 0; c < 4; ++c) oacc[c][vg] *= a;
        }

        // ---- P fragment from LDS in A-layout
        union { v16bf v; uint4 q[2]; } pu;
        const __bf16* pp = &s_p[wv][lane & 15][0] + half;
        pu.q[0] = *(const uint4*)(pp);
        pu.q[1] = *(const uint4*)(pp + 16);

        // ---- O += P V (V^T layout makes B frags contiguous; 4 WMMAs)
        for (int c = 0; c < 4; ++c) {
            v16bf vb = load_frag(vbase, c * 16 + (lane & 15), SEQ, kb0, lane);
            oacc[c] = WMMA_BF16(pu.v, vb, oacc[c]);
        }
    }

    // ---- finalize: O /= l, write bf16 attention output [4096][2048]
    if (lane < 16) s_stat[wv][lane] = 1.0f / l_run;
    lds_fence();
    for (int vg = 0; vg < 8; ++vg) {
        float inv = s_stat[wv][vg + half];
        int qg = q0 + vg + half;
        size_t rowoff = ((size_t)b * SEQ + qg) * DIM + (size_t)h * HD;
        for (int c = 0; c < 4; ++c)
            AO[rowoff + c * 16 + (lane & 15)] = (__bf16)(oacc[c][vg] * inv);
    }
}

// ---------------------------------------------------------------------------
// GEMM3: out = attn_out @ W_out^T  (M=4096, N=2048, K=2048), f32 output
// One wave computes a 32x64 tile.
// ---------------------------------------------------------------------------
__global__ __launch_bounds__(128) void gemm_out_kernel(
        const __bf16* __restrict__ A, const __bf16* __restrict__ W,
        float* __restrict__ out) {
    const int lane = threadIdx.x & 31;
    const int wave = (blockIdx.x * blockDim.x + threadIdx.x) >> 5;
    const int tilesN = DIM / 64;                   // 32
    const int m0 = (wave / tilesN) * 32;
    const int n0 = (wave % tilesN) * 64;

    v8f acc[2][4];
    for (int r = 0; r < 2; ++r)
        for (int c = 0; c < 4; ++c)
            for (int i = 0; i < 8; ++i) acc[r][c][i] = 0.f;

    for (int k0 = 0; k0 < DIM; k0 += 32) {
        v16bf a0 = load_frag(A, m0 +      (lane & 15), DIM, k0, lane);
        v16bf a1 = load_frag(A, m0 + 16 + (lane & 15), DIM, k0, lane);
        v16bf b0 = load_frag(W, n0 +      (lane & 15), DIM, k0, lane);
        v16bf b1 = load_frag(W, n0 + 16 + (lane & 15), DIM, k0, lane);
        v16bf b2 = load_frag(W, n0 + 32 + (lane & 15), DIM, k0, lane);
        v16bf b3 = load_frag(W, n0 + 48 + (lane & 15), DIM, k0, lane);
        acc[0][0] = WMMA_BF16(a0, b0, acc[0][0]);
        acc[1][0] = WMMA_BF16(a1, b0, acc[1][0]);
        acc[0][1] = WMMA_BF16(a0, b1, acc[0][1]);
        acc[1][1] = WMMA_BF16(a1, b1, acc[1][1]);
        acc[0][2] = WMMA_BF16(a0, b2, acc[0][2]);
        acc[1][2] = WMMA_BF16(a1, b2, acc[1][2]);
        acc[0][3] = WMMA_BF16(a0, b3, acc[0][3]);
        acc[1][3] = WMMA_BF16(a1, b3, acc[1][3]);
    }

    const int half = (lane >> 4) << 3;
    for (int r = 0; r < 2; ++r)
        for (int c = 0; c < 4; ++c)
            for (int vg = 0; vg < 8; ++vg) {
                int m = m0 + 16 * r + vg + half;
                int n = n0 + 16 * c + (lane & 15);
                out[(size_t)m * DIM + n] = acc[r][c][vg];
            }
}

// ---------------------------------------------------------------------------
// launch
// ---------------------------------------------------------------------------
extern "C" void kernel_launch(void* const* d_in, const int* in_sizes, int n_in,
                              void* d_out, int out_size, void* d_ws, size_t ws_size,
                              hipStream_t stream) {
    const float* x    = (const float*)d_in[0];
    const float* Wqkv = (const float*)d_in[1];
    const float* Wout = (const float*)d_in[2];
    // d_in[3] (mask) unused: causal -1e9 triu applied inline
    float* out = (float*)d_out;

    char* ws = (char*)d_ws;
    size_t off = 0;
    auto alloc = [&](size_t bytes) -> void* {
        void* p = ws + off;
        off += (bytes + 255) & ~(size_t)255;
        return p;
    };
    __bf16* xb    = (__bf16*)alloc((size_t)MTOT * DIM * 2);
    __bf16* wqkvb = (__bf16*)alloc((size_t)NQKV * DIM * 2);
    __bf16* woutb = (__bf16*)alloc((size_t)DIM * DIM * 2);
    __bf16* Qb    = (__bf16*)alloc((size_t)BATCH * NHEADS   * SEQ * HD * 2);
    __bf16* Kb    = (__bf16*)alloc((size_t)BATCH * NKVHEADS * SEQ * HD * 2);
    __bf16* VTb   = (__bf16*)alloc((size_t)BATCH * NKVHEADS * HD * SEQ * 2);
    __bf16* AOb   = (__bf16*)alloc((size_t)MTOT * DIM * 2);

    f32_to_bf16_kernel<<<(MTOT * DIM + 255) / 256, 256, 0, stream>>>(x, xb, MTOT * DIM);
    f32_to_bf16_kernel<<<(NQKV * DIM + 255) / 256, 256, 0, stream>>>(Wqkv, wqkvb, NQKV * DIM);
    f32_to_bf16_kernel<<<(DIM * DIM + 255) / 256, 256, 0, stream>>>(Wout, woutb, DIM * DIM);

    // QKV projection: (4096/32)*(3072/64) = 6144 waves, 4 waves/block
    gemm_qkv_kernel<<<1536, 128, 0, stream>>>(xb, wqkvb, Qb, Kb, VTb);

    // attention: 2*32*128 = 8192 waves, 8 waves/block
    attn_kernel<<<1024, 256, 0, stream>>>(Qb, Kb, VTb, AOb);

    // output projection: (4096/32)*(2048/64) = 4096 waves
    gemm_out_kernel<<<1024, 128, 0, stream>>>(AOb, woutb, out);
}
